// MultiHead_Attention_41266045780064
// MI455X (gfx1250) — matrix operand order repaired and checked by reading, compiled, not run on hardware
//
#include <hip/hip_runtime.h>
#include <hip/hip_bf16.h>

#define B_  2
#define L_  4096
#define D_  512
#define H_  8
#define HD_ 64
#define SCALE_ 0.125f   // 1/sqrt(64)

typedef __attribute__((ext_vector_type(16))) __bf16 bf16x16;
typedef __attribute__((ext_vector_type(8)))  float  f32x8;
typedef __attribute__((ext_vector_type(4)))  unsigned int u32x4;
typedef __attribute__((ext_vector_type(8)))  int i32x8;
typedef __attribute__((ext_vector_type(4)))  int i32x4;

__device__ __forceinline__ __bf16 f2bf(float x) { return (__bf16)x; }

__device__ __forceinline__ f32x8 wmma_bf16(bf16x16 a, bf16x16 b, f32x8 c) {
  return __builtin_amdgcn_wmma_f32_16x16x32_bf16(
      false, a, false, b, (short)0, c, false, false);
}

// ---- WMMA fragment loaders (CDNA5 ISA 7.12.2 layouts) ----
__device__ __forceinline__ bf16x16 loadA_f32(const float* __restrict__ base,
                                             int row, int ld, int k0, int lane) {
  const float* p = base + row * ld + k0 + ((lane & 16) ? 8 : 0);
  bf16x16 a;
#pragma unroll
  for (int e = 0; e < 8; ++e) { a[e] = f2bf(p[e]); a[8 + e] = f2bf(p[16 + e]); }
  return a;
}
__device__ __forceinline__ bf16x16 loadA_bf16(const __bf16* base,
                                              int row, int ld, int k0, int lane) {
  const __bf16* p = base + row * ld + k0 + ((lane & 16) ? 8 : 0);
  bf16x16 a;
#pragma unroll
  for (int e = 0; e < 8; ++e) { a[e] = p[e]; a[8 + e] = p[16 + e]; }
  return a;
}
__device__ __forceinline__ bf16x16 loadB_bf16(const __bf16* base,
                                              int col, int ld, int k0, int lane) {
  const __bf16* p = base + col * ld + k0 + ((lane & 16) ? 8 : 0);
  bf16x16 b;
#pragma unroll
  for (int e = 0; e < 8; ++e) { b[e] = p[e]; b[8 + e] = p[16 + e]; }
  return b;
}

// ---- TDM: issue a 2D tile load (global -> LDS). 6-arg clang-23 builtin. ----
// D# per CDNA5 ISA: group0 = {flags, lds_addr, gaddr,
// gaddr_hi|type}, group1 = {data_size, tensor dims/tile dims/strides}.
__device__ __forceinline__ void tdm_load_2d(unsigned lds_off,
                                            unsigned long long gaddr,
                                            unsigned dim0, unsigned dim1,
                                            unsigned tile0, unsigned tile1,
                                            unsigned long long stride0) {
  u32x4 g0;
  g0[0] = 1u;                                               // count=1 (valid)
  g0[1] = lds_off;                                          // lds_addr bytes
  g0[2] = (unsigned)(gaddr & 0xFFFFFFFFu);                  // global_addr lo
  g0[3] = (unsigned)((gaddr >> 32) & 0x1FFFFFFu) | (2u << 30);  // hi | type=2
  i32x8 g1;
  g1[0] = (int)(1u << 16);                                  // data_size = 2B
  g1[1] = (int)((dim0 & 0xFFFFu) << 16);                    // tensor_dim0 lo
  g1[2] = (int)(((dim0 >> 16) & 0xFFFFu) | ((dim1 & 0xFFFFu) << 16));
  g1[3] = (int)(((dim1 >> 16) & 0xFFFFu) | (tile0 << 16));  // tile_dim0
  g1[4] = (int)(tile1 & 0xFFFFu);                           // tile_dim1
  g1[5] = (int)(unsigned)(stride0 & 0xFFFFFFFFu);           // dim0_stride lo
  g1[6] = (int)((stride0 >> 32) & 0xFFFFu);                 // dim0_stride hi
  g1[7] = 0;
  i32x4 z4 = {0, 0, 0, 0};
  i32x8 z8 = {0, 0, 0, 0, 0, 0, 0, 0};
  __builtin_amdgcn_tensor_load_to_lds(g0, g1, z4, z4, z8, 0);
}

// ---- Kernel 0: weights fp32 -> bf16 (Wq|Wk|Wv|Wo packed) ----
__global__ void __launch_bounds__(256) cvt_w_kernel(
    const float* __restrict__ Wq, const float* __restrict__ Wk,
    const float* __restrict__ Wv, const float* __restrict__ Wo,
    __bf16* __restrict__ wbf) {
  int tid = blockIdx.x * blockDim.x + threadIdx.x;
  int idx = tid * 4;
  int which = idx / (D_ * D_);
  int off   = idx % (D_ * D_);
  const float* src = (which == 0) ? Wq : (which == 1) ? Wk : (which == 2) ? Wv : Wo;
  float4 v = *(const float4*)(src + off);
  __bf16* dst = wbf + idx;
  dst[0] = f2bf(v.x); dst[1] = f2bf(v.y); dst[2] = f2bf(v.z); dst[3] = f2bf(v.w);
}

// ---- Kernel 1: fused QKV projection (x @ W^T + b) ----
__global__ void __launch_bounds__(256) qkv_proj_kernel(
    const float* __restrict__ xq, const float* __restrict__ xk,
    const float* __restrict__ xv, const __bf16* __restrict__ wbf,
    const float* __restrict__ bq, const float* __restrict__ bk,
    const float* __restrict__ bv,
    __bf16* __restrict__ qh, __bf16* __restrict__ kh, __bf16* __restrict__ vt) {
  const int lane = threadIdx.x & 31;
  const int wave = (blockIdx.x * blockDim.x + threadIdx.x) >> 5;
  const int NSTRIP = D_ / 64;              // 8
  const int MT = (B_ * L_) / 16;           // 512
  int proj = wave / (MT * NSTRIP);
  int rem  = wave % (MT * NSTRIP);
  int mt = rem / NSTRIP;
  int n0 = (rem % NSTRIP) * 64;

  const float*  x    = (proj == 0) ? xq : (proj == 1) ? xk : xv;
  const __bf16* W    = wbf + (size_t)proj * D_ * D_;
  const float*  bias = (proj == 0) ? bq : (proj == 1) ? bk : bv;

  f32x8 acc[4] = {};
  int rowA = mt * 16 + (lane & 15);
  int colB = n0 + (lane & 15);

  for (int k0 = 0; k0 < D_; k0 += 32) {
    bf16x16 a = loadA_f32(x, rowA, D_, k0, lane);
#pragma unroll
    for (int nt = 0; nt < 4; ++nt) {
      bf16x16 b = loadB_bf16(W, colB + nt * 16, D_, k0, lane);
      acc[nt] = wmma_bf16(a, b, acc[nt]);
    }
  }

  int rbase = (lane & 16) ? 8 : 0;
#pragma unroll
  for (int nt = 0; nt < 4; ++nt) {
    int col = colB + nt * 16;
    int h = col >> 6, hd = col & 63;
    float bb = bias[col];
#pragma unroll
    for (int r = 0; r < 8; ++r) {
      int m = mt * 16 + rbase + r;
      int b = m / L_, l = m % L_;
      float v = acc[nt][r] + bb;
      if (proj == 2) {
        vt[(((size_t)b * H_ + h) * HD_ + hd) * L_ + l] = f2bf(v);
      } else {
        __bf16* dst = (proj == 0) ? qh : kh;
        dst[(((size_t)b * H_ + h) * L_ + l) * HD_ + hd] = f2bf(v);
      }
    }
  }
}

// ---- Kernel 2: causal flash attention ----
// Block = 8 waves sharing one (b,h) and a 128-query span; K/V 128-key tiles
// staged to LDS once per block via the Tensor Data Mover; softmax stats
// amortized over 128 keys.
__global__ void __launch_bounds__(256) attn_kernel(
    const __bf16* __restrict__ qh, const __bf16* __restrict__ kh,
    const __bf16* __restrict__ vt, __bf16* __restrict__ ao) {
  __shared__ __align__(16) __bf16 ldsK[128 * HD_];     // [key][hd]   16 KB
  __shared__ __align__(16) __bf16 ldsV[HD_ * 128];     // [hd][key]   16 KB
  __shared__ __align__(16) __bf16 Pshare[8][16 * 64];  // per-wave    16 KB

  const int lane = threadIdx.x & 31;
  const int wib  = threadIdx.x >> 5;
  const int bh     = blockIdx.x >> 5;        // 0..15
  const int qblock = blockIdx.x & 31;        // 0..31 (128 queries each)
  const int q0 = qblock * 128 + wib * 16;

  const __bf16* qhp = qh + (size_t)bh * L_ * HD_;
  const unsigned long long kbase_addr =
      (unsigned long long)(uintptr_t)(kh + (size_t)bh * L_ * HD_);
  const unsigned long long vbase_addr =
      (unsigned long long)(uintptr_t)(vt + (size_t)bh * HD_ * L_);
  const unsigned ldsK_off = (unsigned)(uintptr_t)&ldsK[0];
  const unsigned ldsV_off = (unsigned)(uintptr_t)&ldsV[0];

  bf16x16 aq0 = loadA_bf16(qhp, q0 + (lane & 15), HD_, 0,  lane);
  bf16x16 aq1 = loadA_bf16(qhp, q0 + (lane & 15), HD_, 32, lane);

  const int colL  = lane & 15;
  const int rbase = (lane & 16) ? 8 : 0;

  float mrow[8], lrow[8];
#pragma unroll
  for (int r = 0; r < 8; ++r) { mrow[r] = -1e30f; lrow[r] = 0.f; }
  f32x8 acc[4] = {};

  for (int kb = 0; kb <= qblock; ++kb) {     // 128-key blocks, uniform count
    int kbg = kb * 128;
    if (wib == 0) {
      // K tile: 128 consecutive rows of [L,HD] -> flat 16 KB copy
      tdm_load_2d(ldsK_off, kbase_addr + (size_t)kbg * HD_ * 2,
                  128 * HD_, 1, 128 * HD_, 1, 128 * HD_);
      // V tile: 64 rows (hd) x 128 keys from [HD,L] (row stride L)
      tdm_load_2d(ldsV_off, vbase_addr + (size_t)kbg * 2,
                  L_, HD_, 128, HD_, L_);
      __builtin_amdgcn_s_wait_tensorcnt(0);
    }
    __syncthreads();                         // K/V tiles visible to all waves

    int last = q0 + 15 - kbg;                // last needed local key
    int nsub = (last < 0) ? 0 : ((last >> 5) + 1);
    if (nsub > 4) nsub = 4;                  // 32-key subtiles this wave

    // ---- scores: up to 8 D-tiles (16 cols each) from LDS K ----
    f32x8 S[8];
#pragma unroll
    for (int t = 0; t < 4; ++t) {
      if (t < nsub) {
        f32x8 s0 = {}, s1 = {};
        bf16x16 b0 = loadB_bf16(ldsK, t * 32 + colL, HD_, 0,  lane);
        bf16x16 b1 = loadB_bf16(ldsK, t * 32 + colL, HD_, 32, lane);
        s0 = wmma_bf16(aq0, b0, s0);
        s0 = wmma_bf16(aq1, b1, s0);
        bf16x16 b2 = loadB_bf16(ldsK, t * 32 + 16 + colL, HD_, 0,  lane);
        bf16x16 b3 = loadB_bf16(ldsK, t * 32 + 16 + colL, HD_, 32, lane);
        s1 = wmma_bf16(aq0, b2, s1);
        s1 = wmma_bf16(aq1, b3, s1);
        S[2 * t] = s0; S[2 * t + 1] = s1;
      }
    }

    // ---- one softmax update per 128 keys ----
    if (nsub > 0) {
#pragma unroll
      for (int r = 0; r < 8; ++r) {
        int qrow = q0 + rbase + r;
        float mloc = -1e30f;
#pragma unroll
        for (int tt = 0; tt < 8; ++tt) {
          if (tt < 2 * nsub) {
            int key = kbg + tt * 16 + colL;
            float s = (key <= qrow) ? S[tt][r] * SCALE_ : -1e30f;
            S[tt][r] = s;
            mloc = fmaxf(mloc, s);
          }
        }
#pragma unroll
        for (int m = 1; m <= 8; m <<= 1) mloc = fmaxf(mloc, __shfl_xor(mloc, m, 32));
        float mnew  = fmaxf(mrow[r], mloc);
        float alpha = __expf(mrow[r] - mnew);
        float ps = 0.f;
#pragma unroll
        for (int tt = 0; tt < 8; ++tt) {
          if (tt < 2 * nsub) {
            float p = __expf(S[tt][r] - mnew);
            S[tt][r] = p;                    // keep P in the S registers
            ps += p;
          }
        }
#pragma unroll
        for (int m = 1; m <= 8; m <<= 1) ps += __shfl_xor(ps, m, 32);
        lrow[r] = lrow[r] * alpha + ps;
        mrow[r] = mnew;
#pragma unroll
        for (int nt = 0; nt < 4; ++nt) acc[nt][r] *= alpha;
      }

      // ---- PV in two 64-key halves via a 2KB per-wave LDS P buffer ----
#pragma unroll
      for (int half = 0; half < 2; ++half) {
        if (half * 2 < nsub) {
#pragma unroll
          for (int tt = 0; tt < 4; ++tt) {           // 16-col tiles of half
            if (half * 4 + tt < 2 * nsub) {
#pragma unroll
              for (int r = 0; r < 8; ++r)
                Pshare[wib][(rbase + r) * 64 + tt * 16 + colL] =
                    f2bf(S[half * 4 + tt][r]);
            }
          }
          asm volatile("s_wait_dscnt 0" ::: "memory");  // wave-local fence
#pragma unroll
          for (int c = 0; c < 2; ++c) {               // 32-key chunks
            if (half * 2 + c < nsub) {
              bf16x16 pa = loadA_bf16(&Pshare[wib][0], lane & 15, 64,
                                      c * 32, lane);
#pragma unroll
              for (int nt = 0; nt < 4; ++nt) {
                bf16x16 bv = loadB_bf16(ldsV, nt * 16 + colL, 128,
                                        (half * 2 + c) * 32, lane);
                acc[nt] = wmma_bf16(pa, bv, acc[nt]);
              }
            }
          }
        }
      }
    }
    __syncthreads();                         // protect K/V before next stage
  }

  int b = bh / H_, h = bh % H_;
#pragma unroll
  for (int nt = 0; nt < 4; ++nt) {
#pragma unroll
    for (int r = 0; r < 8; ++r) {
      int qrow = q0 + rbase + r;
      float o = acc[nt][r] / lrow[r];
      ao[((size_t)b * L_ + qrow) * D_ + h * HD_ + nt * 16 + colL] = f2bf(o);
    }
  }
}

// ---- Kernel 3: output projection (ao @ Wo^T + bo) -> fp32 out ----
__global__ void __launch_bounds__(256) out_proj_kernel(
    const __bf16* __restrict__ ao, const __bf16* __restrict__ wo,
    const float* __restrict__ bo, float* __restrict__ out) {
  const int lane = threadIdx.x & 31;
  const int wave = (blockIdx.x * blockDim.x + threadIdx.x) >> 5;
  const int NSTRIP = D_ / 64;
  int mt = wave / NSTRIP;
  int n0 = (wave % NSTRIP) * 64;

  f32x8 acc[4] = {};
  int rowA = mt * 16 + (lane & 15);
  int colB = n0 + (lane & 15);

  for (int k0 = 0; k0 < D_; k0 += 32) {
    bf16x16 a = loadA_bf16(ao, rowA, D_, k0, lane);
#pragma unroll
    for (int nt = 0; nt < 4; ++nt) {
      bf16x16 b = loadB_bf16(wo, colB + nt * 16, D_, k0, lane);
      acc[nt] = wmma_bf16(a, b, acc[nt]);
    }
  }

  int rbase = (lane & 16) ? 8 : 0;
#pragma unroll
  for (int nt = 0; nt < 4; ++nt) {
    int col = colB + nt * 16;
    float bb = bo[col];
#pragma unroll
    for (int r = 0; r < 8; ++r) {
      out[(size_t)(mt * 16 + rbase + r) * D_ + col] = acc[nt][r] + bb;
    }
  }
}

extern "C" void kernel_launch(void* const* d_in, const int* in_sizes, int n_in,
                              void* d_out, int out_size, void* d_ws, size_t ws_size,
                              hipStream_t stream) {
  const float* q  = (const float*)d_in[0];
  const float* k  = (const float*)d_in[1];
  const float* v  = (const float*)d_in[2];
  // d_in[3] = causal mask (computed analytically in-kernel)
  const float* Wq = (const float*)d_in[4];
  const float* bq = (const float*)d_in[5];
  const float* Wk = (const float*)d_in[6];
  const float* bk = (const float*)d_in[7];
  const float* Wv = (const float*)d_in[8];
  const float* bv = (const float*)d_in[9];
  const float* Wo = (const float*)d_in[10];
  const float* bo = (const float*)d_in[11];
  float* out = (float*)d_out;

  char* ws = (char*)d_ws;
  __bf16* qh  = (__bf16*)(ws);                          //  8 MiB [B,H,L,HD]
  __bf16* kh  = (__bf16*)(ws + (((size_t)8)  << 20));   //  8 MiB [B,H,L,HD]
  __bf16* vt  = (__bf16*)(ws + (((size_t)16) << 20));   //  8 MiB [B,H,HD,L]
  __bf16* ao  = (__bf16*)(ws + (((size_t)24) << 20));   //  8 MiB [B,L,D]
  __bf16* wbf = (__bf16*)(ws + (((size_t)32) << 20));   //  2 MiB Wq|Wk|Wv|Wo

  cvt_w_kernel<<<1024, 256, 0, stream>>>(Wq, Wk, Wv, Wo, wbf);
  qkv_proj_kernel<<<1536, 256, 0, stream>>>(q, k, v, wbf, bq, bk, bv, qh, kh, vt);
  // 16 (b,h) * 32 query-blocks of 128 = 512 blocks x 8 waves
  attn_kernel<<<512, 256, 0, stream>>>(qh, kh, vt, ao);
  out_proj_kernel<<<512, 256, 0, stream>>>(ao, wbf + (size_t)3 * D_ * D_, bo, out);
}
